// SparseAutoencoder_27152783245520
// MI455X (gfx1250) — matrix operand order repaired and checked by reading, not compiled, not run on hardware
//
#include <hip/hip_runtime.h>
#include <hip/hip_bf16.h>

// ---------- problem constants ----------
static constexpr int D_MODEL = 1024;
static constexpr int D_SAE   = 16384;
static constexpr int BATCH   = 4096;
static constexpr int TOP_K   = 32;
static constexpr unsigned K_TOTAL = TOP_K * BATCH;   // 131072 kept entries

// ---------- vector types ----------
typedef __attribute__((ext_vector_type(16))) __bf16         v16bf;
typedef __attribute__((ext_vector_type(8)))  float          v8f;
typedef __attribute__((ext_vector_type(8)))  float          float8;
typedef __attribute__((ext_vector_type(16))) unsigned short ushort16;
typedef __attribute__((ext_vector_type(8)))  unsigned short ushort8;
typedef __attribute__((ext_vector_type(4)))  int            int4v;

// gfx1250 async global->LDS path (guarded: falls back to load+ds_store)
#if defined(__HIP_DEVICE_COMPILE__) && \
    __has_builtin(__builtin_amdgcn_global_load_async_to_lds_b128) && \
    __has_builtin(__builtin_amdgcn_s_wait_asynccnt)
#define HAVE_GFX1250_ASYNC 1
#else
#define HAVE_GFX1250_ASYNC 0
#endif

__device__ __forceinline__ void async_copy_b128(const unsigned short* g, unsigned short* l) {
#if HAVE_GFX1250_ASYNC
    __builtin_amdgcn_global_load_async_to_lds_b128(
        (__attribute__((address_space(1))) int4v*)g,
        (__attribute__((address_space(3))) int4v*)l, 0, 0);
#else
    *(ushort8*)l = *(const ushort8*)g;
#endif
}

__device__ __forceinline__ void wait_async_copies() {
#if HAVE_GFX1250_ASYNC
    __builtin_amdgcn_s_wait_asynccnt(0);
#endif
}

// round-to-nearest-even f32 -> bf16 (bit trick; inputs are finite)
__device__ __forceinline__ unsigned short f2bf(float f) {
    unsigned u = __float_as_uint(f);
    u += 0x7FFFu + ((u >> 16) & 1u);
    return (unsigned short)(u >> 16);
}

// =====================================================================
// Transpose + convert: in fp32 [R x C] row-major  ->  out bf16 [C x R]
// =====================================================================
__global__ __launch_bounds__(256) void transpose_f32_to_bf16(
    const float* __restrict__ in, unsigned short* __restrict__ out,
    int R, int C)
{
    __shared__ float tile[32][33];
    const int c0 = blockIdx.x * 32;
    const int r0 = blockIdx.y * 32;
    const int tx = threadIdx.x;   // 0..31
    const int ty = threadIdx.y;   // 0..7
#pragma unroll
    for (int i = ty; i < 32; i += 8)
        tile[i][tx] = in[(size_t)(r0 + i) * C + (c0 + tx)];
    __syncthreads();
#pragma unroll
    for (int i = ty; i < 32; i += 8)
        out[(size_t)(c0 + i) * R + (r0 + tx)] = f2bf(tile[tx][i]);
}

// =====================================================================
// LDS-blocked WMMA GEMM:
//   C[M,N] = A[M,K](fp32, converted bf16 on the fly) x Bt[N,K](bf16) + bias[N]
// Block = 256 threads = 8 waves; block tile 128M x 128N; K stepped by 32.
// Wave grid 2(M) x 4(N); each wave owns 64M x 32N = 8 WMMA tiles.
// B tile staged via GLOBAL_LOAD_ASYNC_TO_LDS_B128 (ASYNCcnt), A staged
// through VGPRs with f32->bf16 conversion. Optional fused |value|
// histogram (top-k pass 1) over the outputs, reusing staging LDS.
// =====================================================================
__global__ __launch_bounds__(256) void wmma_gemm_bf16_blocked(
    const float*          __restrict__ A,
    const unsigned short* __restrict__ Bt,
    const float*          __restrict__ bias,
    float*                __restrict__ C,
    unsigned*             __restrict__ gh1,   // nullptr: no histogram
    int M, int N, int K)
{
    __shared__ __align__(32) unsigned char smemraw[16384];
    unsigned short* ldsA = (unsigned short*)smemraw;           // 128 x 32 bf16 (8KB)
    unsigned short* ldsB = (unsigned short*)(smemraw + 8192);  // 128 x 32 bf16 (8KB)

    const int tid  = threadIdx.x;
    const int lane = tid & 31;
    const int wave = tid >> 5;
    const int bn_tiles = N >> 7;
    const int bm = blockIdx.x / bn_tiles;
    const int bn = blockIdx.x - bm * bn_tiles;
    const int m0 = bm << 7;
    const int n0 = bn << 7;

    const int  wm = wave >> 2;    // 0..1 (M)
    const int  wn = wave & 3;     // 0..3 (N)
    const int  lr = lane & 15;
    const bool hi = lane >= 16;
    const int  kofsA = hi ? 16 : 0;
    const int  kofsB = hi ? 16 : 0;

    // ---- staging geometry ----
    // A: thread t covers row t/2, K-halves of 16; two float8 loads + ushort8 stores.
    const int ar = tid >> 1;
    const int ah = (tid & 1) << 4;
    const float* Ag = A + (size_t)(m0 + ar) * K + ah;
    unsigned short* Al = ldsA + ar * 32 + ah;
    // B: 512 16B chunks; thread t handles chunks t and t+256.
    const int br = tid >> 2;
    const int bq = (tid & 3) << 3;
    const unsigned short* Bg0 = Bt + (size_t)(n0 + br)      * K + bq;
    const unsigned short* Bg1 = Bt + (size_t)(n0 + br + 64) * K + bq;
    unsigned short* Bl0 = ldsB + br * 32 + bq;
    unsigned short* Bl1 = ldsB + (br + 64) * 32 + bq;

    v8f acc[4][2] = {};

    for (int kk = 0; kk < K; kk += 32) {
        // ---- stage B tile (async global -> LDS) ----
        async_copy_b128(Bg0 + kk, Bl0);
        async_copy_b128(Bg1 + kk, Bl1);
        // ---- stage A tile (f32 load, convert, LDS store) ----
        float8 fa = *(const float8*)(Ag + kk);
        float8 fb = *(const float8*)(Ag + kk + 8);
        __builtin_prefetch((const void*)(Ag + kk + 32), 0, 1);
        __builtin_prefetch((const void*)(Bg0 + kk + 32), 0, 1);
        ushort8 ua, ub;
#pragma unroll
        for (int j = 0; j < 8; ++j) { ua[j] = f2bf(fa[j]); ub[j] = f2bf(fb[j]); }
        *(ushort8*)(Al)     = ua;
        *(ushort8*)(Al + 8) = ub;

        wait_async_copies();
        __syncthreads();

        // ---- B fragments (32B contiguous per lane) ----
        v16bf b0 = __builtin_bit_cast(v16bf,
            *(const ushort16*)(ldsB + (wn * 32 + lr) * 32 + kofsB));
        v16bf b1 = __builtin_bit_cast(v16bf,
            *(const ushort16*)(ldsB + (wn * 32 + 16 + lr) * 32 + kofsB));

        // ---- A fragments + 8 WMMAs ----
#pragma unroll
        for (int Mi = 0; Mi < 4; ++Mi) {
            const int abase = (wm * 64 + Mi * 16 + lr) * 32 + kofsA;
            ushort8 alo = *(const ushort8*)(ldsA + abase);
            ushort8 ahi = *(const ushort8*)(ldsA + abase + 8);
            v16bf af = __builtin_bit_cast(v16bf,
                __builtin_shufflevector(alo, ahi,
                    0, 1, 2, 3, 4, 5, 6, 7, 8, 9, 10, 11, 12, 13, 14, 15));
            acc[Mi][0] = __builtin_amdgcn_wmma_f32_16x16x32_bf16(
                false, af, false, b0, (short)0, acc[Mi][0], false, false);
            acc[Mi][1] = __builtin_amdgcn_wmma_f32_16x16x32_bf16(
                false, af, false, b1, (short)0, acc[Mi][1], false, false);
        }
        __syncthreads();
    }

    // ---- optional fused histogram pass 1 (reuse staging LDS as 4096 bins) ----
    unsigned* lh = (unsigned*)smemraw;
    if (gh1 != nullptr) {
        for (int i = tid; i < 4096; i += 256) lh[i] = 0;
        __syncthreads();
    }

    // ---- epilogue: bias add + store (+ histogram) ----
    const int mbase = m0 + wm * 64 + (hi ? 8 : 0);
    const int nbase = n0 + wn * 32;
#pragma unroll
    for (int Mi = 0; Mi < 4; ++Mi) {
#pragma unroll
        for (int Nj = 0; Nj < 2; ++Nj) {
            const int col = nbase + Nj * 16 + lr;
            const float bv = bias[col];
#pragma unroll
            for (int r = 0; r < 8; ++r) {
                float v = acc[Mi][Nj][r] + bv;
                C[(size_t)(mbase + Mi * 16 + r) * N + col] = v;
                if (gh1 != nullptr)
                    atomicAdd(&lh[(__float_as_uint(v) & 0x7FFFFFFFu) >> 19], 1u);
            }
        }
    }
    if (gh1 != nullptr) {
        __syncthreads();
        for (int i = tid; i < 4096; i += 256)
            if (lh[i]) atomicAdd(&gh1[i], lh[i]);
    }
}

// =====================================================================
// Global abs-top-k threshold: two-level 12-bit radix histograms over the
// monotonic |float| bit pattern (pass 1 fused into encode GEMM).
// =====================================================================
__global__ void select1_kernel(const unsigned* __restrict__ gh,
                               unsigned* __restrict__ sel, unsigned kwant)
{
    if (threadIdx.x != 0 || blockIdx.x != 0) return;
    unsigned long long cum = 0;
    unsigned b = 0, need = kwant;
    for (int i = 4095; i >= 0; --i) {
        unsigned long long c2 = cum + gh[i];
        if (c2 >= kwant) { b = (unsigned)i; need = (unsigned)(kwant - cum); break; }
        cum = c2;
    }
    sel[0] = b; sel[1] = need;
}

__global__ __launch_bounds__(256) void hist2_kernel(
    const float* __restrict__ h, const unsigned* __restrict__ sel,
    unsigned* __restrict__ gh2, size_t n)
{
    __shared__ unsigned lh[4096];
    for (int i = threadIdx.x; i < 4096; i += blockDim.x) lh[i] = 0;
    __syncthreads();
    const unsigned b1 = sel[0];
    const size_t stride = (size_t)gridDim.x * blockDim.x;
    for (size_t i = (size_t)blockIdx.x * blockDim.x + threadIdx.x; i < n; i += stride) {
        unsigned u = __float_as_uint(h[i]) & 0x7FFFFFFFu;
        if ((u >> 19) == b1) atomicAdd(&lh[(u >> 7) & 0xFFFu], 1u);
    }
    __syncthreads();
    for (int i = threadIdx.x; i < 4096; i += blockDim.x)
        if (lh[i]) atomicAdd(&gh2[i], lh[i]);
}

__global__ void select2_kernel(const unsigned* __restrict__ gh2,
                               unsigned* __restrict__ sel)
{
    if (threadIdx.x != 0 || blockIdx.x != 0) return;
    const unsigned b1 = sel[0];
    const unsigned need = sel[1];
    unsigned long long cum = 0;
    unsigned b2 = 0;
    for (int i = 4095; i >= 0; --i) {
        cum += gh2[i];
        if (cum >= need) { b2 = (unsigned)i; break; }
    }
    sel[2] = (b1 << 19) | (b2 << 7);   // keep if |bits| >= threshold
}

__global__ __launch_bounds__(256) void mask_kernel(
    float* __restrict__ fx, const unsigned* __restrict__ sel, size_t n)
{
    const unsigned T = sel[2];
    const size_t stride = (size_t)gridDim.x * blockDim.x;
    for (size_t i = (size_t)blockIdx.x * blockDim.x + threadIdx.x; i < n; i += stride) {
        float v = fx[i];
        unsigned u = __float_as_uint(v) & 0x7FFFFFFFu;
        if (u < T) fx[i] = 0.0f;
    }
}

// =====================================================================
// Launch
// =====================================================================
extern "C" void kernel_launch(void* const* d_in, const int* in_sizes, int n_in,
                              void* d_out, int out_size, void* d_ws, size_t ws_size,
                              hipStream_t stream) {
    (void)in_sizes; (void)n_in; (void)out_size; (void)ws_size;

    const float* x     = (const float*)d_in[0];   // [BATCH, D_MODEL]
    const float* W_enc = (const float*)d_in[1];   // [D_MODEL, D_SAE]
    const float* b_enc = (const float*)d_in[2];   // [D_SAE]
    const float* W_dec = (const float*)d_in[3];   // [D_SAE, D_MODEL]
    const float* b_dec = (const float*)d_in[4];   // [D_MODEL]

    float* x_hat = (float*)d_out;                                   // [BATCH, D_MODEL]
    float* f_x   = (float*)d_out + (size_t)BATCH * D_MODEL;         // [BATCH, D_SAE]

    // workspace layout: 64MB transposed bf16 weights + histograms
    char* ws = (char*)d_ws;
    unsigned short* WencT = (unsigned short*)ws;                                  // bf16 [D_SAE, D_MODEL]
    unsigned short* WdecT = (unsigned short*)(ws + (size_t)D_SAE * D_MODEL * 2);  // bf16 [D_MODEL, D_SAE]
    unsigned* gh1 = (unsigned*)(ws + (size_t)D_SAE * D_MODEL * 4);
    unsigned* gh2 = gh1 + 4096;
    unsigned* sel = gh2 + 4096;

    (void)hipMemsetAsync(gh1, 0, (4096 * 2 + 8) * sizeof(unsigned), stream);

    transpose_f32_to_bf16<<<dim3(D_SAE / 32, D_MODEL / 32), dim3(32, 8), 0, stream>>>(
        W_enc, WencT, D_MODEL, D_SAE);
    transpose_f32_to_bf16<<<dim3(D_MODEL / 32, D_SAE / 32), dim3(32, 8), 0, stream>>>(
        W_dec, WdecT, D_SAE, D_MODEL);

    // encode: h_pre -> f_x region, with fused |value| histogram (top-k pass 1)
    {
        const int blocks = (BATCH / 128) * (D_SAE / 128);   // 32 * 128 = 4096
        wmma_gemm_bf16_blocked<<<blocks, 256, 0, stream>>>(
            x, WencT, b_enc, f_x, gh1, BATCH, D_SAE, D_MODEL);
    }

    // global abs top-k threshold + mask (in place on f_x)
    const size_t nh = (size_t)BATCH * D_SAE;
    select1_kernel<<<1, 32, 0, stream>>>(gh1, sel, K_TOTAL);
    hist2_kernel<<<2048, 256, 0, stream>>>(f_x, sel, gh2, nh);
    select2_kernel<<<1, 32, 0, stream>>>(gh2, sel);
    mask_kernel<<<2048, 256, 0, stream>>>(f_x, sel, nh);

    // decode: x_hat = f_x @ W_dec + b_dec
    {
        const int blocks = (BATCH / 128) * (D_MODEL / 128); // 32 * 8 = 256
        wmma_gemm_bf16_blocked<<<blocks, 256, 0, stream>>>(
            f_x, WdecT, b_dec, x_hat, nullptr, BATCH, D_MODEL, D_SAE);
    }
}
